// MultiHeadAttention_21509196218874
// MI455X (gfx1250) — matrix operand order repaired and checked by reading, not compiled, not run on hardware
//
#include <hip/hip_runtime.h>

// ---------------------------------------------------------------------------
// MHA with Shaw relative position embeddings on gfx1250 (MI455X).
// - All GEMM work through v_wmma_f32_16x16x32_f16 (f16 operands, f32 accum).
// - Attention K/V tiles staged to LDS with global_load_async_to_lds_b128
//   (ASYNCcnt / s_wait_asynccnt), double-buffered, shared by all 8 waves.
// Compute-bound workload (~69 GFLOP GEMM, ~150MB traffic vs 23.3 TB/s HBM).
// ---------------------------------------------------------------------------

typedef __attribute__((ext_vector_type(16))) _Float16 v16h;
typedef __attribute__((ext_vector_type(8)))  _Float16 v8h;
typedef __attribute__((ext_vector_type(8)))  float    v8f;

#define B_     2
#define L_     2048
#define E_     1024
#define H_     16
#define HD_    64
#define M_ROWS 4096   // B_*L_

static __device__ inline v8f wmma_f16(v16h a, v16h b, v8f c) {
  return __builtin_amdgcn_wmma_f32_16x16x32_f16(
      false, a, false, b, (short)0, c, false, false);
}

// A-operand (16x32 f16) per-lane load. Contiguous k layout (matches B):
//   lanes 0-15  (kh=0): row M=lane,    K = {0..15}
//   lanes 16-31 (kh=1): row M=lane-16, K = {16..31}
static __device__ inline v16h load_A16(const _Float16* rowptr, int kbase, int kh) {
  union { v16h v; v8h h[2]; } u;
  u.h[0] = *(const v8h*)(rowptr + kbase + kh * 16);
  u.h[1] = *(const v8h*)(rowptr + kbase + kh * 16 + 8);
  return u.v;
}

// Async global->LDS 16B per lane (gfx1250 GLOBAL_LOAD_ASYNC_TO_LDS_B128).
// First operand: VGPR holding LDS byte address (generic LDS addr low 32 bits);
// second: 64-bit global address; tracked by ASYNCcnt.
static __device__ inline void async_cp16(void* lds_dst, const void* gsrc) {
  unsigned int loff = (unsigned int)(unsigned long long)lds_dst;
  asm volatile("global_load_async_to_lds_b128 %0, %1, off"
               :: "v"(loff), "v"(gsrc) : "memory");
}
#define S_WAIT_ASYNC(n) asm volatile("s_wait_asynccnt " #n ::: "memory")

// ---------------------------------------------------------------------------
// f32 -> f16 conversion (grid-stride)
// ---------------------------------------------------------------------------
__global__ void cvt_f32_f16(const float* __restrict__ src,
                            _Float16* __restrict__ dst, int n) {
  int i = blockIdx.x * blockDim.x + threadIdx.x;
  int stride = gridDim.x * blockDim.x;
  for (; i < n; i += stride) dst[i] = (_Float16)src[i];
}

// ---------------------------------------------------------------------------
// Projection GEMM: Out = relu(X @ W^T + bias), X:[M][K] f16, W:[N][K] f16.
// Wave tile 32(M) x 64(N): 8 WMMAs per K-step for 12 b128 loads.
// mode 0: head-split Out[b][h][l][d]; mode 1: transposed Out[b][h][d][l].
// ---------------------------------------------------------------------------
__global__ __launch_bounds__(256) void proj_gemm(
    const _Float16* __restrict__ X, const _Float16* __restrict__ W,
    const float* __restrict__ bias, _Float16* __restrict__ Out,
    int K, int mode) {
  const int lane = threadIdx.x & 31;
  const int w    = threadIdx.x >> 5;
  const int n16  = lane & 15;
  const int kh   = lane >> 4;
  const int mbase = blockIdx.y * 256 + w * 32;
  const int nbase = blockIdx.x * 64;

  const _Float16* arow0 = X + (size_t)(mbase + n16) * K;
  const _Float16* arow1 = X + (size_t)(mbase + 16 + n16) * K;

  v8f acc[2][4];
#pragma unroll
  for (int mm = 0; mm < 2; ++mm)
#pragma unroll
    for (int t = 0; t < 4; ++t)
#pragma unroll
      for (int j = 0; j < 8; ++j) acc[mm][t][j] = 0.f;

  for (int k0 = 0; k0 < K; k0 += 32) {
    v16h a0 = load_A16(arow0, k0, kh);
    v16h a1 = load_A16(arow1, k0, kh);
#pragma unroll
    for (int t = 0; t < 4; ++t) {
      const _Float16* wrow = W + (size_t)(nbase + t * 16 + n16) * K;
      v16h b = *(const v16h*)(wrow + k0 + kh * 16);
      acc[0][t] = wmma_f16(a0, b, acc[0][t]);
      acc[1][t] = wmma_f16(a1, b, acc[1][t]);
    }
  }

#pragma unroll
  for (int mm = 0; mm < 2; ++mm) {
#pragma unroll
    for (int t = 0; t < 4; ++t) {
#pragma unroll
      for (int j = 0; j < 8; ++j) {
        int m   = mbase + mm * 16 + j + 8 * kh;
        int col = nbase + t * 16 + n16;
        float v = acc[mm][t][j] + bias[col];
        v = v > 0.f ? v : 0.f;
        int b_ = m >> 11, l_ = m & (L_ - 1);
        int h_ = col >> 6, d_ = col & (HD_ - 1);
        size_t off;
        if (mode == 0)
          off = (((size_t)(b_ * H_ + h_) * L_) + l_) * HD_ + d_;
        else
          off = (((size_t)(b_ * H_ + h_) * HD_) + d_) * L_ + l_;
        Out[off] = (_Float16)v;
      }
    }
  }
}

// ---------------------------------------------------------------------------
// Output GEMM: d_out = relu(Ymid @ Wo^T + bo), f32 output, row-major.
// ---------------------------------------------------------------------------
__global__ __launch_bounds__(256) void out_gemm(
    const _Float16* __restrict__ X, const _Float16* __restrict__ W,
    const float* __restrict__ bias, float* __restrict__ Out, int K, int N) {
  const int lane = threadIdx.x & 31;
  const int w    = threadIdx.x >> 5;
  const int n16  = lane & 15;
  const int kh   = lane >> 4;
  const int mbase = blockIdx.y * 256 + w * 32;
  const int nbase = blockIdx.x * 64;

  const _Float16* arow0 = X + (size_t)(mbase + n16) * K;
  const _Float16* arow1 = X + (size_t)(mbase + 16 + n16) * K;

  v8f acc[2][4];
#pragma unroll
  for (int mm = 0; mm < 2; ++mm)
#pragma unroll
    for (int t = 0; t < 4; ++t)
#pragma unroll
      for (int j = 0; j < 8; ++j) acc[mm][t][j] = 0.f;

  for (int k0 = 0; k0 < K; k0 += 32) {
    v16h a0 = load_A16(arow0, k0, kh);
    v16h a1 = load_A16(arow1, k0, kh);
#pragma unroll
    for (int t = 0; t < 4; ++t) {
      const _Float16* wrow = W + (size_t)(nbase + t * 16 + n16) * K;
      v16h b = *(const v16h*)(wrow + k0 + kh * 16);
      acc[0][t] = wmma_f16(a0, b, acc[0][t]);
      acc[1][t] = wmma_f16(a1, b, acc[1][t]);
    }
  }

#pragma unroll
  for (int mm = 0; mm < 2; ++mm) {
#pragma unroll
    for (int t = 0; t < 4; ++t) {
#pragma unroll
      for (int j = 0; j < 8; ++j) {
        int m   = mbase + mm * 16 + j + 8 * kh;
        int col = nbase + t * 16 + n16;
        float v = acc[mm][t][j] + bias[col];
        Out[(size_t)m * N + col] = v > 0.f ? v : 0.f;
      }
    }
  }
}

// ---------------------------------------------------------------------------
// P[bh][q][j] = sum_d Qh[bh][q][d] * rel_k[j][d],  j = 0..32
// ---------------------------------------------------------------------------
__global__ void relp_kernel(const _Float16* __restrict__ Qh,
                            const float* __restrict__ rel_k,
                            float* __restrict__ P) {
  int idx = blockIdx.x * blockDim.x + threadIdx.x;
  const int total = B_ * H_ * L_ * 33;
  if (idx >= total) return;
  int j  = idx % 33;
  int qi = idx / 33;  // bh*L + q
  const _Float16* qp = Qh + (size_t)qi * HD_;
  const float* rp = rel_k + j * HD_;
  float s = 0.f;
#pragma unroll 8
  for (int d = 0; d < HD_; ++d) s += (float)qp[d] * rp[d];
  P[idx] = s;
}

// ---------------------------------------------------------------------------
// Flash attention with Shaw relative bias.
// Grid: (16 q-tiles of 128, 32 bh). Block: 256 = 8 waves; wave owns 16 q rows.
// K/V tiles double-buffered in LDS via global_load_async_to_lds_b128
// (one async instruction per wave per tile; all 8 waves share the tiles).
// ---------------------------------------------------------------------------
__global__ __launch_bounds__(256) void attn_kernel(
    const _Float16* __restrict__ Qh, const _Float16* __restrict__ Kh,
    const _Float16* __restrict__ Vt, const float* __restrict__ P,
    const float* __restrict__ rel_v, _Float16* __restrict__ Ymid) {
  __shared__ float relv_s[33][64];
  __shared__ float PrWb[8][16][34];   // rel-bias rows during loop, buckets after
  __shared__ alignas(32) _Float16 Pt[8][16][32];
  __shared__ alignas(32) _Float16 Kst[2][32][64];  // [buf][k_local][d]
  __shared__ alignas(32) _Float16 Vst[2][64][32];  // [buf][d][k_local]

  const int lane = threadIdx.x & 31;
  const int w    = threadIdx.x >> 5;
  const int n16  = lane & 15;
  const int hi   = lane >> 4;
  const int bh   = blockIdx.y;
  const int qbase = blockIdx.x * 128 + w * 16;
  const int tid  = threadIdx.x;

  for (int t = tid; t < 33 * 64; t += 256)
    relv_s[t / 64][t % 64] = rel_v[t];
  for (int t = lane; t < 16 * 33; t += 32) {
    int r = t / 33, c = t % 33;
    PrWb[w][r][c] = P[((size_t)bh * L_ + qbase + r) * 33 + c];
  }

  // cooperative async staging of one 32-k group (K: contiguous 4KB; V: 64x64B)
  auto issue_group = [&](int g, int buf) {
    const char* gk = (const char*)(Kh + ((size_t)bh * L_ + g * 32) * HD_);
    async_cp16((char*)&Kst[buf][0][0] + tid * 16, gk + tid * 16);
    int row = tid >> 2, c = tid & 3;
    const _Float16* gv = Vt + ((size_t)bh * HD_ + row) * L_ + g * 32 + c * 8;
    async_cp16(&Vst[buf][row][c * 8], gv);
  };
  issue_group(0, 0);
  __syncthreads();  // relv_s / PrWb staging visible

  const _Float16* qrow = Qh + ((size_t)bh * L_ + qbase + n16) * HD_;
  v16h Aq0 = load_A16(qrow, 0, hi);
  v16h Aq1 = load_A16(qrow, 32, hi);

  float mrun[8], lrun[8], wlo[8], whi[8], pn0[8], pn1[8];
  int qrowj[8];
  v8f O[4];
#pragma unroll
  for (int t = 0; t < 4; ++t)
#pragma unroll
    for (int j = 0; j < 8; ++j) O[t][j] = 0.f;
#pragma unroll
  for (int j = 0; j < 8; ++j) {
    mrun[j] = -1e30f; lrun[j] = 0.f;
    wlo[j] = whi[j] = pn0[j] = pn1[j] = 0.f;
    qrowj[j] = qbase + j + 8 * hi;
  }
  const float scale = 0.03125f;  // 1/sqrt(D_KEY=1024)
  const int G = L_ / 32;

  for (int g = 0; g < G; ++g) {
    const int buf = g & 1;
    const int k0 = g * 32;
    if (g + 1 < G) { issue_group(g + 1, buf ^ 1); S_WAIT_ASYNC(2); }
    else           { S_WAIT_ASYNC(0); }
    __syncthreads();  // group g fully staged by all waves

    float s8[2][8];
#pragma unroll
    for (int h2 = 0; h2 < 2; ++h2) {
      const _Float16* kr = &Kst[buf][h2 * 16 + n16][0];
      v16h B0 = *(const v16h*)(kr + hi * 16);        // d 0..31
      v16h B1 = *(const v16h*)(kr + 32 + hi * 16);   // d 32..63
      v8f S;
#pragma unroll
      for (int j = 0; j < 8; ++j) S[j] = 0.f;
      S = wmma_f16(Aq0, B0, S);
      S = wmma_f16(Aq1, B1, S);
#pragma unroll
      for (int j = 0; j < 8; ++j) {
        int d   = k0 + h2 * 16 + n16 - qrowj[j];
        int bkt = d < -16 ? 0 : (d > 16 ? 32 : d + 16);
        s8[h2][j] = (S[j] + PrWb[w][j + 8 * hi][bkt]) * scale;
      }
    }
    float corr[8];
#pragma unroll
    for (int j = 0; j < 8; ++j) {
      float v = fmaxf(s8[0][j], s8[1][j]);
      v = fmaxf(v, __shfl_xor(v, 1, 32));
      v = fmaxf(v, __shfl_xor(v, 2, 32));
      v = fmaxf(v, __shfl_xor(v, 4, 32));
      v = fmaxf(v, __shfl_xor(v, 8, 32));
      float mn = fmaxf(mrun[j], v);
      corr[j] = __expf(mrun[j] - mn);
      mrun[j] = mn;
    }
    float p0[8], p1[8];
#pragma unroll
    for (int j = 0; j < 8; ++j) {
      p0[j] = __expf(s8[0][j] - mrun[j]);
      p1[j] = __expf(s8[1][j] - mrun[j]);
    }
#pragma unroll
    for (int j = 0; j < 8; ++j) {
      float v = p0[j] + p1[j];
      v += __shfl_xor(v, 1, 32);
      v += __shfl_xor(v, 2, 32);
      v += __shfl_xor(v, 4, 32);
      v += __shfl_xor(v, 8, 32);
      lrun[j] = lrun[j] * corr[j] + v;
    }
#pragma unroll
    for (int t = 0; t < 4; ++t)
#pragma unroll
      for (int j = 0; j < 8; ++j) O[t][j] *= corr[j];
#pragma unroll
    for (int j = 0; j < 8; ++j) {
      wlo[j] *= corr[j]; whi[j] *= corr[j];
      pn0[j] *= corr[j]; pn1[j] *= corr[j];
    }
#pragma unroll
    for (int h2 = 0; h2 < 2; ++h2) {
#pragma unroll
      for (int j = 0; j < 8; ++j) {
        float p = h2 ? p1[j] : p0[j];
        int d = k0 + h2 * 16 + n16 - qrowj[j];
        if (d <= -16)      wlo[j] += p;
        else if (d >= 16)  whi[j] += p;
        else if (d < 0)    pn0[j] += p;   // bucket d+16 in 1..15
        else               pn1[j] += p;   // bucket d+16 in 16..31
      }
    }
#pragma unroll
    for (int j = 0; j < 8; ++j) {
      Pt[w][j + 8 * hi][n16]      = (_Float16)p0[j];
      Pt[w][j + 8 * hi][16 + n16] = (_Float16)p1[j];
    }
    __syncthreads();
    v16h Ap;
    {
      union { v16h v; v8h h[2]; } u;
      u.h[0] = *(const v8h*)(&Pt[w][n16][hi * 16]);
      u.h[1] = *(const v8h*)(&Pt[w][n16][hi * 16 + 8]);
      Ap = u.v;
    }
#pragma unroll
    for (int t = 0; t < 4; ++t) {
      v16h Bv = *(const v16h*)(&Vst[buf][t * 16 + n16][hi * 16]);
      O[t] = wmma_f16(Ap, Bv, O[t]);
    }
    __syncthreads();  // all waves done with buf before g+2 overwrites it
  }

  // bucket weights w[q, 0..32]: reuse PrWb (rel-bias lifetime ended)
  for (int t = lane; t < 16 * 34; t += 32) PrWb[w][t / 34][t % 34] = 0.f;
  __syncthreads();
#pragma unroll
  for (int j = 0; j < 8; ++j) {
    int m = j + 8 * hi;
    int r = (n16 - qrowj[j]) & 15;
    float a = wlo[j], b2 = whi[j];
    a  += __shfl_xor(a, 1, 32);  a  += __shfl_xor(a, 2, 32);
    a  += __shfl_xor(a, 4, 32);  a  += __shfl_xor(a, 8, 32);
    b2 += __shfl_xor(b2, 1, 32); b2 += __shfl_xor(b2, 2, 32);
    b2 += __shfl_xor(b2, 4, 32); b2 += __shfl_xor(b2, 8, 32);
    if (r > 0) PrWb[w][m][r] = pn0[j];
    PrWb[w][m][r + 16] = pn1[j];
    if (n16 == 0) { PrWb[w][m][0] = a; PrWb[w][m][32] = b2; }
  }
  __syncthreads();

  const int b_ = bh >> 4, h_ = bh & 15;
#pragma unroll
  for (int t = 0; t < 4; ++t) {
#pragma unroll
    for (int j = 0; j < 8; ++j) {
      int m = j + 8 * hi;
      int dcol = t * 16 + n16;
      float rs = 0.f;
      for (int jj = 0; jj < 33; ++jj) rs += PrWb[w][m][jj] * relv_s[jj][dcol];
      float val = (O[t][j] + rs) / lrun[j];
      size_t off = (((size_t)b_ * L_ + qbase + m) * E_) + h_ * HD_ + dcol;
      Ymid[off] = (_Float16)val;
    }
  }
}

// ---------------------------------------------------------------------------
// Launch
// ---------------------------------------------------------------------------
extern "C" void kernel_launch(void* const* d_in, const int* in_sizes, int n_in,
                              void* d_out, int out_size, void* d_ws, size_t ws_size,
                              hipStream_t stream) {
  const float* q     = (const float*)d_in[0];
  const float* k     = (const float*)d_in[1];
  const float* v     = (const float*)d_in[2];
  const float* Wq    = (const float*)d_in[3];
  const float* bq    = (const float*)d_in[4];
  const float* Wk    = (const float*)d_in[5];
  const float* bk    = (const float*)d_in[6];
  const float* Wv    = (const float*)d_in[7];
  const float* bv    = (const float*)d_in[8];
  const float* Wo    = (const float*)d_in[9];
  const float* bo    = (const float*)d_in[10];
  const float* rel_k = (const float*)d_in[11];
  const float* rel_v = (const float*)d_in[12];

  char* ws = (char*)d_ws;
  size_t off = 0;
  auto alloc = [&](size_t bytes) -> char* {
    char* p = ws + off;
    off += (bytes + 255) & ~size_t(255);
    return p;
  };
  const size_t actN = (size_t)M_ROWS * E_;
  const size_t wN   = (size_t)E_ * E_;
  _Float16* q16  = (_Float16*)alloc(actN * 2);
  _Float16* k16  = (_Float16*)alloc(actN * 2);
  _Float16* v16  = (_Float16*)alloc(actN * 2);
  _Float16* Wq16 = (_Float16*)alloc(wN * 2);
  _Float16* Wk16 = (_Float16*)alloc(wN * 2);
  _Float16* Wv16 = (_Float16*)alloc(wN * 2);
  _Float16* Wo16 = (_Float16*)alloc(wN * 2);
  _Float16* Qh   = (_Float16*)alloc(actN * 2);   // [b][h][l][d]
  _Float16* Kh   = (_Float16*)alloc(actN * 2);   // [b][h][l][d]
  _Float16* Vt   = (_Float16*)alloc(actN * 2);   // [b][h][d][l]
  _Float16* Ymid = (_Float16*)alloc(actN * 2);   // [b][l][e]
  float*    Pbuf = (float*)   alloc((size_t)B_ * H_ * L_ * 33 * 4);

  cvt_f32_f16<<<2048, 256, 0, stream>>>(q,  q16,  (int)actN);
  cvt_f32_f16<<<2048, 256, 0, stream>>>(k,  k16,  (int)actN);
  cvt_f32_f16<<<2048, 256, 0, stream>>>(v,  v16,  (int)actN);
  cvt_f32_f16<<<1024, 256, 0, stream>>>(Wq, Wq16, (int)wN);
  cvt_f32_f16<<<1024, 256, 0, stream>>>(Wk, Wk16, (int)wN);
  cvt_f32_f16<<<1024, 256, 0, stream>>>(Wv, Wv16, (int)wN);
  cvt_f32_f16<<<1024, 256, 0, stream>>>(Wo, Wo16, (int)wN);

  dim3 pgrid(E_ / 64, M_ROWS / 256);  // (16, 16)
  proj_gemm<<<pgrid, 256, 0, stream>>>(q16, Wq16, bq, Qh, E_, 0);
  proj_gemm<<<pgrid, 256, 0, stream>>>(k16, Wk16, bk, Kh, E_, 0);
  proj_gemm<<<pgrid, 256, 0, stream>>>(v16, Wv16, bv, Vt, E_, 1);

  {
    int total = B_ * H_ * L_ * 33;
    relp_kernel<<<(total + 255) / 256, 256, 0, stream>>>(Qh, rel_k, Pbuf);
  }

  dim3 agrid(L_ / 128, B_ * H_);  // (16, 32)
  attn_kernel<<<agrid, 256, 0, stream>>>(Qh, Kh, Vt, Pbuf, rel_v, Ymid);

  out_gemm<<<pgrid, 256, 0, stream>>>(Ymid, Wo16, bo, (float*)d_out, E_, E_);
}
